// TFN_6373731467315
// MI455X (gfx1250) — matrix operand order repaired and checked by reading, not compiled, not run on hardware
//
#include <hip/hip_runtime.h>
#include <hip/hip_bf16.h>

#define NE 8192
#define NN 2048

typedef __attribute__((ext_vector_type(16))) _Float16 v16h;
typedef __attribute__((ext_vector_type(8)))  _Float16 v8h;
typedef __attribute__((ext_vector_type(8)))  float    v8f;
typedef _Float16 H;

__host__ __device__ inline int imin_(int a, int b){ return a < b ? a : b; }
__host__ __device__ inline int iabs_(int a){ return a < 0 ? -a : a; }

// ---- offsets into the Q (CG) table and basis buffer --------------------
__host__ __device__ inline int qpair_off(int di, int dd){
  int off = 0;
  for(int a = 0; a < 4; ++a) for(int b = 0; b < 4; ++b){
    if(a == di && b == dd) return off;
    int lo = iabs_(a - b);
    for(int J = lo; J <= a + b; ++J) off += (2*a+1)*(2*J+1)*(2*b+1);
  }
  return off;
}
__host__ __device__ inline int bpair_off(int di, int dd){
  int off = 0;
  for(int a = 0; a < 4; ++a) for(int b = 0; b < 4; ++b){
    if(a == di && b == dd) return off;
    off += (2*a+1)*(2*b+1)*(2*imin_(a,b)+1);
  }
  return off;
}
// total basis floats per edge = 1092; total Q floats = 7056

// ---- device CG machinery ------------------------------------------------
struct cx { double re, im; };
__device__ inline cx cmul(cx a, cx b){ cx r; r.re = a.re*b.re - a.im*b.im; r.im = a.re*b.im + a.im*b.re; return r; }

__device__ double dfact(int n){ double r = 1.0; for(int i = 2; i <= n; ++i) r *= (double)i; return r; }

__device__ double su2cg(int j1,int m1,int j2,int m2,int j3,int m3){
  if(m1 + m2 != m3) return 0.0;
  if(m1 < -j1 || m1 > j1 || m2 < -j2 || m2 > j2 || m3 < -j3 || m3 > j3) return 0.0;
  double pre = sqrt((2.0*j3+1.0)*dfact(j1+j2-j3)*dfact(j1-j2+j3)*dfact(-j1+j2+j3)/dfact(j1+j2+j3+1));
  pre *= sqrt(dfact(j1+m1)*dfact(j1-m1)*dfact(j2+m2)*dfact(j2-m2)*dfact(j3+m3)*dfact(j3-m3));
  double s = 0.0;
  for(int k = 0; k <= j1+j2-j3; ++k){
    int t1 = j1+j2-j3-k, t2 = j1-m1-k, t3 = j2+m2-k, t4 = j3-j2+m1+k, t5 = j3-j1-m2+k;
    if(t1 < 0 || t2 < 0 || t3 < 0 || t4 < 0 || t5 < 0) continue;
    double d = dfact(k)*dfact(t1)*dfact(t2)*dfact(t3)*dfact(t4)*dfact(t5);
    s += ((k & 1) ? -1.0 : 1.0) / d;
  }
  return pre * s;
}

__device__ void c2rU(int l, cx* U){
  int n = 2*l+1;
  for(int i = 0; i < n*n; ++i){ U[i].re = 0.0; U[i].im = 0.0; }
  U[l*n+l].re = 1.0;
  double s2 = sqrt(0.5);
  for(int m = 1; m <= l; ++m){
    double sg = (m & 1) ? -1.0 : 1.0;
    U[(l+m)*n+(l+m)].re = sg*s2;
    U[(l+m)*n+(l-m)].re = s2;
    U[(l-m)*n+(l-m)].im = s2;
    U[(l-m)*n+(l+m)].im = -sg*s2;
  }
}

// out layout: [i(2l1+1)][m(2l2+1)][o(2l3+1)]
__device__ void real_cg_dev(int l1, int l2, int l3, float* out){
  cx U1[49], U2[169], U3[49];
  c2rU(l1, U1); c2rU(l2, U2); c2rU(l3, U3);
  int n1 = 2*l1+1, n2 = 2*l2+1, n3 = 2*l3+1;
  double Tre[637], Tim[637];
  double sr = 0.0, si = 0.0;
  for(int a = 0; a < n1; ++a) for(int c = 0; c < n2; ++c) for(int e = 0; e < n3; ++e){
    double ar = 0.0, ai = 0.0;
    for(int bb = 0; bb < n1; ++bb) for(int d = 0; d < n2; ++d) for(int f = 0; f < n3; ++f){
      double cg = su2cg(l1, bb-l1, l2, d-l2, l3, f-l3);
      if(cg == 0.0) continue;
      cx u = cmul(U1[a*n1+bb], U2[c*n2+d]);
      cx v; v.re = U3[e*n3+f].re; v.im = -U3[e*n3+f].im;
      cx w = cmul(u, v);
      ar += w.re*cg; ai += w.im*cg;
    }
    int id = (a*n2+c)*n3+e;
    Tre[id] = ar; Tim[id] = ai;
    sr += fabs(ar); si += fabs(ai);
  }
  int tot = n1*n2*n3;
  for(int i = 0; i < tot; ++i) out[i] = (float)((sr >= si) ? Tre[i] : Tim[i]);
}

__global__ void k_cg(float* Q){
  int t = threadIdx.x;
  if(t >= 16) return;
  int di = t >> 2, dd = t & 3;
  int lo = iabs_(di - dd);
  float* out = Q + qpair_off(di, dd);
  for(int J = lo; J <= di + dd; ++J){
    real_cg_dev(di, J, dd, out);
    out += (2*di+1)*(2*J+1)*(2*dd+1);
  }
}

// ---- edge geometry: spherical harmonics up to l=6 ----------------------
__device__ float ffact(int n){ float r = 1.f; for(int i = 2; i <= n; ++i) r *= (float)i; return r; }

__global__ void k_geom(const float* x, const float* ea, const int* ei, float* Y, float* ef){
  int e = blockIdx.x*blockDim.x + threadIdx.x;
  if(e >= NE) return;
  int s = ei[e], d = ei[NE + e];
  float dx = x[s*3+0]-x[d*3+0], dy = x[s*3+1]-x[d*3+1], dz = x[s*3+2]-x[d*3+2];
  float r = sqrtf(dx*dx + dy*dy + dz*dz + 1e-12f);
  float xx = dx/r, yy = dy/r, zz = dz/r;
  float phi = atan2f(yy, xx);
  float st = sqrtf(fmaxf(0.f, fminf(1.f, 1.f - zz*zz)));
  float P[7][7];
  P[0][0] = 1.f;
  for(int m = 1; m <= 6; ++m) P[m][m] = -(float)(2*m-1)*st*P[m-1][m-1];
  for(int m = 0; m < 6; ++m) P[m+1][m] = (float)(2*m+1)*zz*P[m][m];
  for(int m = 0; m <= 6; ++m)
    for(int l = m+2; l <= 6; ++l)
      P[l][m] = ((float)(2*l-1)*zz*P[l-1][m] - (float)(l-1+m)*P[l-2][m]) / (float)(l-m);
  const float TWOPI = 6.28318530717958648f;
  float* Ye = Y + (size_t)e*49;
  for(int l = 0; l <= 6; ++l){
    float* Yl = Ye + l*l;
    Yl[l] = sqrtf((float)(2*l+1)/(2.f*TWOPI))*P[l][0];
    for(int m = 1; m <= l; ++m){
      float nl = sqrtf((float)(2*l+1)/TWOPI * ffact(l-m)/ffact(l+m));
      Yl[l-m] = nl*sinf((float)m*phi)*P[l][m];
      Yl[l+m] = nl*cosf((float)m*phi)*P[l][m];
    }
  }
  ef[e*5+0] = ea[e*4+0]; ef[e*5+1] = ea[e*4+1];
  ef[e*5+2] = ea[e*4+2]; ef[e*5+3] = ea[e*4+3];
  ef[e*5+4] = r;
}

// ---- basis: B[e,o,i,j] = sum_m Q[i,m,o] * Y_J[e,m] ---------------------
__global__ void k_basis(const float* Q, const float* Y, float* B){
  int e = blockIdx.x*blockDim.x + threadIdx.x;
  if(e >= NE) return;
  const float* Ye = Y + (size_t)e*49;
  for(int di = 0; di < 4; ++di) for(int dd = 0; dd < 4; ++dd){
    int ni = 2*di+1, no = 2*dd+1, lo = iabs_(di-dd), nf = 2*imin_(di,dd)+1;
    const float* Qp = Q + qpair_off(di, dd);
    float* Bp = B + (size_t)NE*bpair_off(di, dd) + (size_t)e*no*ni*nf;
    for(int j = 0; j < nf; ++j){
      int J = lo + j, nm = 2*J+1;
      const float* Yl = Ye + J*J;
      for(int o = 0; o < no; ++o) for(int i = 0; i < ni; ++i){
        float acc = 0.f;
        for(int m = 0; m < nm; ++m) acc += Qp[(i*nm+m)*no+o]*Yl[m];
        Bp[(o*ni+i)*nf + j] = acc;
      }
      Qp += ni*nm*no;
    }
  }
}

// ---- small utility kernels ---------------------------------------------
__global__ void k_zero(float* p, int n){
  int i = blockIdx.x*blockDim.x + threadIdx.x;
  if(i < n) p[i] = 0.f;
}
__global__ void k_count(const int* dst, float* cnt){
  int e = blockIdx.x*blockDim.x + threadIdx.x;
  if(e < NE) atomicAdd(&cnt[dst[e]], 1.f);
}
__global__ void k_deg(const float* cnt, float* deg){
  int n = blockIdx.x*blockDim.x + threadIdx.x;
  if(n < NN) deg[n] = 1.f/fmaxf(cnt[n], 1.f);
}
__global__ void k_cvt(const float* w, H* o, int n){
  int i = blockIdx.x*blockDim.x + threadIdx.x;
  if(i < n) o[i] = (H)w[i];
}

// ---- radial MLP ---------------------------------------------------------
__global__ void k_lin1(const float* ef, const float* W1, const float* b1, float* out){
  int idx = blockIdx.x*blockDim.x + threadIdx.x;
  if(idx >= NE*32) return;
  int e = idx >> 5, c = idx & 31;
  const float* f = ef + e*5;
  float acc = b1[c];
  for(int k = 0; k < 5; ++k) acc += f[k]*W1[k*32+c];
  out[idx] = acc;
}

__global__ void k_bnstats(const float* a, float* stats){
  __shared__ float s1[256], s2[256];
  int c = blockIdx.x, t = threadIdx.x;
  float su = 0.f, sq = 0.f;
  for(int e = t; e < NE; e += 256){ float v = a[e*32+c]; su += v; sq += v*v; }
  s1[t] = su; s2[t] = sq; __syncthreads();
  for(int o = 128; o > 0; o >>= 1){
    if(t < o){ s1[t] += s1[t+o]; s2[t] += s2[t+o]; }
    __syncthreads();
  }
  if(t == 0){ float mu = s1[0]/(float)NE; stats[c] = mu; stats[32+c] = s2[0]/(float)NE - mu*mu; }
}

__global__ void k_bnrelu(const float* a, const float* stats, const float* g, const float* be, H* out){
  int idx = blockIdx.x*blockDim.x + threadIdx.x;
  if(idx >= NE*32) return;
  int c = idx & 31;
  float hh = (a[idx]-stats[c])*rsqrtf(stats[32+c]+1e-5f)*g[c] + be[c];
  out[idx] = (H)(hh > 0.f ? hh : 0.f);
}

// ---- WMMA helpers (contiguous k order, same in A and B, wave32) ------
// A fragment of a row-major [16 x 32] f16 tile, lda halves between rows.
// lanes 0-15: row=lane,   K in {0..15}
// lanes 16-31: row=lane-16, K in {16..31}
__device__ inline v16h load_afrag(const H* base, int lda, int lane){
  int row = lane & 15;
  int kb = (lane & 16) ? 16 : 0;
  const H* p = base + (size_t)row*lda;
  v8h lo = *(const v8h*)(p + kb);
  v8h hi = *(const v8h*)(p + kb + 8);
  v16h a;
  for(int i = 0; i < 8; ++i){ a[i] = lo[i]; a[8+i] = hi[i]; }
  return a;
}

// ---- WMMA GEMM (generic tile): C[E x M] = A[E x 32] * W[32 x M] + bias -
__global__ void k_gemm1(const H* __restrict__ A, const H* __restrict__ B,
                        const float* __restrict__ bias, float* __restrict__ C, int M){
  int l = threadIdx.x;
  int e0 = blockIdx.x * 16;
  int n0 = blockIdx.y * 16;

  __builtin_prefetch(A + (size_t)(e0 + 16)*32, 0, 1);   // next edge tile -> global_prefetch_b8
  v16h a = load_afrag(A + (size_t)e0*32, 32, l);

  int colB = l & 15;
  int kbB = (l & 16) ? 16 : 0;
  const H* Bp = B + (size_t)n0 + colB;
  v16h b;
  for(int i = 0; i < 16; ++i) b[i] = Bp[(size_t)(kbB + i)*M];

  float bv = bias ? bias[n0 + colB] : 0.f;
  v8f c;
  for(int i = 0; i < 8; ++i) c[i] = bv;

  c = __builtin_amdgcn_wmma_f32_16x16x32_f16(false, a, false, b, (short)0, c, false, false);

  int mb = (l & 16) ? 8 : 0;
  float* Cp = C + (size_t)e0*M + n0 + colB;
  for(int v = 0; v < 8; ++v) Cp[(size_t)(mb + v)*M] = c[v];
}

// ---- WMMA GEMM, LDS-staged W tile, 8 waves x 4 N-tiles per block -------
// Block: 256 threads = 8 waves; covers 128 edges x 64 output columns.
// W tile [32 x 64] staged once in LDS (4 KB), shared by all 8 waves.
__global__ void k_gemm4(const H* __restrict__ A, const H* __restrict__ B,
                        const float* __restrict__ bias, float* __restrict__ C, int M){
  __shared__ H Bs[32*64];
  int t = threadIdx.x;
  int n0 = blockIdx.y * 64;
  {
    int idx = t*8, row = idx >> 6, col = idx & 63;
    *(v8h*)&Bs[row*64 + col] = *(const v8h*)(B + (size_t)row*M + n0 + col);
  }
  __syncthreads();

  int l = t & 31, wid = t >> 5;
  int e0 = blockIdx.x*128 + wid*16;
  v16h a = load_afrag(A + (size_t)e0*32, 32, l);

  int colB = l & 15;
  int kbB = (l & 16) ? 16 : 0;
  int mb  = (l & 16) ? 8 : 0;
#pragma unroll
  for(int tt = 0; tt < 4; ++tt){
    v16h b;
    for(int i = 0; i < 16; ++i) b[i] = Bs[(kbB + i)*64 + tt*16 + colB];
    float bv = bias[n0 + tt*16 + colB];
    v8f c;
    for(int i = 0; i < 8; ++i) c[i] = bv;
    c = __builtin_amdgcn_wmma_f32_16x16x32_f16(false, a, false, b, (short)0, c, false, false);
    float* Cp = C + (size_t)e0*M + n0 + tt*16 + colB;
    for(int v = 0; v < 8; ++v) Cp[(size_t)(mb + v)*M] = c[v];
  }
}

// ---- T = einsum('eoij,eci->ecoj', basis, h_src) ------------------------
__global__ void k_T(const float* Bp_, const float* feat, const int* src, float* T,
                    int ic, int ni, int no, int nf){
  int idx = blockIdx.x*blockDim.x + threadIdx.x;
  if(idx >= NE*ic) return;
  int e = idx/ic, c = idx - e*ic;
  int s = src[e];
  const float* hp = feat + ((size_t)s*ic + c)*ni;
  float hv[7];
  for(int i = 0; i < ni; ++i) hv[i] = hp[i];
  const float* Bp = Bp_ + (size_t)e*no*ni*nf;
  float* Tp = T + ((size_t)e*ic + c)*no*nf;
  for(int o = 0; o < no; ++o) for(int j = 0; j < nf; ++j){
    float acc = 0.f;
    for(int i = 0; i < ni; ++i) acc += Bp[(o*ni+i)*nf+j]*hv[i];
    Tp[o*nf+j] = acc;
  }
}

// ---- msg[e,k,o] += sum_{c,j} R[e,k,c,j] * T[e,c,o,j] -------------------
__global__ void k_msg(const float* R, const float* T, float* msg, int oc, int ic, int nf, int no){
  int idx = blockIdx.x*blockDim.x + threadIdx.x;
  if(idx >= NE*oc) return;
  int e = idx/oc, k = idx - e*oc;
  size_t M = (size_t)oc*ic*nf;
  const float* Rp = R + (size_t)e*M + (size_t)k*ic*nf;
  const float* Tp = T + (size_t)e*ic*no*nf;
  float acc[7];
  for(int o = 0; o < no; ++o) acc[o] = 0.f;
  for(int c = 0; c < ic; ++c) for(int j = 0; j < nf; ++j){
    float rv = Rp[c*nf+j];
    for(int o = 0; o < no; ++o) acc[o] += rv*Tp[(c*no+o)*nf+j];
  }
  float* mp = msg + ((size_t)e*oc + k)*no;
  for(int o = 0; o < no; ++o) mp[o] += acc[o];
}

__global__ void k_scatter(const float* msg, const int* dst, float* agg, int oc, int no){
  int idx = blockIdx.x*blockDim.x + threadIdx.x;
  int tot = NE*oc*no;
  if(idx >= tot) return;
  int e = idx/(oc*no), r = idx - e*oc*no;
  atomicAdd(&agg[(size_t)dst[e]*oc*no + r], msg[idx]);
}

__global__ void k_finish(const float* agg, const float* deg, const float* S, const float* fin,
                         float* fout, int oc, int no, int icin){
  int idx = blockIdx.x*blockDim.x + threadIdx.x;
  int tot = NN*oc*no;
  if(idx >= tot) return;
  int n = idx/(oc*no), r = idx - n*oc*no;
  int k = r/no, o = r - k*no;
  float v = agg[idx]*deg[n];
  if(S){
    const float* fp = fin + (size_t)n*icin*no + o;
    float acc = 0.f;
    for(int i = 0; i < icin; ++i) acc += S[k*icin+i]*fp[(size_t)i*no];
    v += acc;
  }
  fout[idx] = v;
}

// ---- equivariant norm (in-place per node) ------------------------------
__global__ void k_gnorm(float* f, const float* g, const float* b, const float* W, const float* cb, int nm){
  int n = blockIdx.x*blockDim.x + threadIdx.x;
  if(n >= NN) return;
  float* fp = f + (size_t)n*16*nm;
  float nrm[16], hv[16], outh[16];
  float mu = 0.f;
  for(int c = 0; c < 16; ++c){
    float s = 0.f;
    for(int m = 0; m < nm; ++m){ float xv = fp[c*nm+m]; s += xv*xv; }
    nrm[c] = sqrtf(s + 1e-12f);
    mu += nrm[c];
  }
  mu *= (1.f/16.f);
  float var = 0.f;
  for(int c = 0; c < 16; ++c){ float d = nrm[c]-mu; var += d*d; }
  var *= (1.f/16.f);
  float inv = rsqrtf(var + 1e-5f);
  for(int c = 0; c < 16; ++c){
    float hh = (nrm[c]-mu)*inv*g[c] + b[c];
    hv[c] = hh > 0.f ? hh : 0.f;
  }
  for(int k = 0; k < 16; ++k){
    float acc = cb[k];
    for(int c = 0; c < 16; ++c) acc += hv[c]*W[c*16+k];
    outh[k] = acc;
  }
  for(int k = 0; k < 16; ++k){
    float sc = outh[k]/nrm[k];
    for(int m = 0; m < nm; ++m) fp[k*nm+m] *= sc;
  }
}

// ---- output heads -------------------------------------------------------
// h_pred = feats0[NN x 64] @ Wmap[64 x 16] + bmap   (K=64 -> 2 chained WMMAs)
__global__ void k_head_h(const H* __restrict__ Af, const H* __restrict__ Wm,
                         const float* __restrict__ bm, float* __restrict__ out){
  int l = threadIdx.x;
  int n0 = blockIdx.x * 16;
  int colB = l & 15;
  int kbB = (l & 16) ? 16 : 0;
  float bv = bm[colB];
  v8f c;
  for(int i = 0; i < 8; ++i) c[i] = bv;
#pragma unroll
  for(int kt = 0; kt < 2; ++kt){
    v16h a = load_afrag(Af + (size_t)n0*64 + kt*32, 64, l);
    v16h b;
    for(int i = 0; i < 16; ++i) b[i] = Wm[(size_t)(kt*32 + kbB + i)*16 + colB];
    c = __builtin_amdgcn_wmma_f32_16x16x32_f16(false, a, false, b, (short)0, c, false, false);
  }
  int mb = (l & 16) ? 8 : 0;
  float* Cp = out + (size_t)n0*16 + colB;
  for(int v = 0; v < 8; ++v) Cp[(size_t)(mb + v)*16] = c[v];
}
__global__ void k_head_x(const float* f1, const float* x, float* out){
  int idx = blockIdx.x*blockDim.x + threadIdx.x;
  if(idx >= NN*3) return;
  int n = idx/3, c = idx - n*3;
  out[idx] = f1[(size_t)n*2*3 + c] + x[idx];
}

// ========================================================================
extern "C" void kernel_launch(void* const* d_in, const int* in_sizes, int n_in,
                              void* d_out, int out_size, void* d_ws, size_t ws_size,
                              hipStream_t stream){
  const float* hin = (const float*)d_in[0];
  const float* x   = (const float*)d_in[1];
  const float* ea  = (const float*)d_in[2];
  const int*   ei  = (const int*)d_in[3];

  // ---- walk params in pytree-flatten (sorted-key) order ----
  int pi = 4;
  auto P = [&]() -> const float* { return (const float*)d_in[pi++]; };
  const float* Wmap = P();
  const float* bmap = P();
  struct Rad { const float *W1,*W2,*W3,*b1,*b2,*b3,*be1,*be2,*g1,*g2; };
  Rad rad[4][4][4];                       // [layer][di][do]
  const float* smat[4][4] = {{0}};        // [layer][do]
  const float *nG[3][4], *nB[3][4], *nW[3][4], *nC[3][4];
  for(int li = 0; li < 4; ++li){
    int nDi = (li == 0) ? 1 : 4;
    int nDo = (li == 3) ? 2 : 4;
    for(int a = 0; a < nDi; ++a) for(int b = 0; b < nDo; ++b){
      Rad& r = rad[li][a][b];
      r.W1 = P(); r.W2 = P(); r.W3 = P();
      r.b1 = P(); r.b2 = P(); r.b3 = P();
      r.be1 = P(); r.be2 = P(); r.g1 = P(); r.g2 = P();
    }
    for(int b = 0; b < nDo; ++b){
      bool inFin = (li == 0) ? (b == 0) : true;
      if(inFin) smat[li][b] = P();
    }
    if(li < 3){
      for(int d = 0; d < 4; ++d) nW[li][d] = P();
      for(int d = 0; d < 4; ++d) nB[li][d] = P();
      for(int d = 0; d < 4; ++d) nC[li][d] = P();
      for(int d = 0; d < 4; ++d) nG[li][d] = P();
    }
  }

  // ---- workspace layout (float units) ----
  float* w = (float*)d_ws;
  const size_t E = NE, Nn = NN;
  size_t oQ   = 0;
  size_t oY   = oQ + 7056;
  size_t oEf  = oY + E*49;
  size_t oCnt = oEf + E*5;
  size_t oDeg = oCnt + Nn;
  size_t oB   = oDeg + Nn;
  size_t oFa  = oB + E*1092;
  size_t oFb  = oFa + 4*Nn*448;
  size_t oA1  = oFb + 4*Nn*448;
  size_t oA2  = oA1 + E*32;
  size_t oH1  = oA2 + E*32;       // E*32 halves
  size_t oH2  = oH1 + E*16;       // E*32 halves
  size_t oW2h = oH2 + E*16;       // 1024 halves
  size_t oW3h = oW2h + 512;       // up to 57344 halves
  size_t oSt  = oW3h + 28672;     // 64 floats of BN stats
  size_t oMsg = oSt + 64;         // E*112
  size_t oAgg = oMsg + E*112;     // Nn*448
  size_t oT   = oAgg + Nn*448;    // E*784
  size_t oR   = oT + E*784;       // E*1792

  float* a1   = w + oA1;
  float* a2   = w + oA2;
  H*     h1   = (H*)(w + oH1);
  H*     h2   = (H*)(w + oH2);
  H*     W2h  = (H*)(w + oW2h);
  H*     W3h  = (H*)(w + oW3h);
  float* st   = w + oSt;
  float* msgb = w + oMsg;
  float* aggb = w + oAgg;
  float* Tb   = w + oT;
  float* Rb   = w + oR;
  float* cnt  = w + oCnt;
  float* deg  = w + oDeg;

  auto zero = [&](float* p, int n){
    k_zero<<<(n+255)/256, 256, 0, stream>>>(p, n);
  };

  // ---- setup: CG tables, geometry, basis, degree ----
  k_cg   <<<1, 16, 0, stream>>>(w + oQ);
  k_geom <<<NE/256, 256, 0, stream>>>(x, ea, ei, w + oY, w + oEf);
  k_basis<<<(NE+127)/128, 128, 0, stream>>>(w + oQ, w + oY, w + oB);
  zero(cnt, NN);
  k_count<<<NE/256, 256, 0, stream>>>(ei + NE, cnt);
  k_deg  <<<NN/256, 256, 0, stream>>>(cnt, deg);

  float* FAp[4]; float* FBp[4];
  for(int d = 0; d < 4; ++d){
    FAp[d] = w + oFa + (size_t)d*Nn*448;
    FBp[d] = w + oFb + (size_t)d*Nn*448;
  }
  const float* featIn[4] = {hin, 0, 0, 0};

  for(int li = 0; li < 4; ++li){
    bool first = (li == 0), last = (li == 3);
    float** outBuf = (li & 1) ? FBp : FAp;
    int nDo = last ? 2 : 4;
    for(int b = 0; b < nDo; ++b){
      int dd = b;
      int oc = last ? (dd == 0 ? 64 : 2) : 16;
      int no = 2*dd + 1;
      zero(msgb, NE*oc*no);
      int nDi = first ? 1 : 4;
      for(int di = 0; di < nDi; ++di){
        int ic = first ? 32 : 16;
        const Rad& rp = rad[li][di][dd];
        // radial MLP
        k_lin1   <<<NE*32/256, 256, 0, stream>>>(w + oEf, rp.W1, rp.b1, a1);
        k_bnstats<<<32, 256, 0, stream>>>(a1, st);
        k_bnrelu <<<NE*32/256, 256, 0, stream>>>(a1, st, rp.g1, rp.be1, h1);
        k_cvt    <<<4, 256, 0, stream>>>(rp.W2, W2h, 1024);
        k_gemm1  <<<dim3(NE/16, 2), 32, 0, stream>>>(h1, W2h, rp.b2, a2, 32);
        k_bnstats<<<32, 256, 0, stream>>>(a2, st);
        k_bnrelu <<<NE*32/256, 256, 0, stream>>>(a2, st, rp.g2, rp.be2, h2);
        int nf = 2*imin_(di, dd) + 1;
        int M = nf*ic*oc;
        k_cvt    <<<(32*M+255)/256, 256, 0, stream>>>(rp.W3, W3h, 32*M);
        if((M & 63) == 0)
          k_gemm4<<<dim3(NE/128, M/64), 256, 0, stream>>>(h2, W3h, rp.b3, Rb, M);
        else
          k_gemm1<<<dim3(NE/16, M/16), 32, 0, stream>>>(h2, W3h, rp.b3, Rb, M);
        // basis application + message
        int ni = 2*di + 1;
        const float* Bpair = w + oB + (size_t)NE*bpair_off(di, dd);
        const float* fsrc = first ? hin : featIn[di];
        k_T  <<<(NE*ic+255)/256, 256, 0, stream>>>(Bpair, fsrc, ei, Tb, ic, ni, no, nf);
        k_msg<<<(NE*oc+255)/256, 256, 0, stream>>>(Rb, Tb, msgb, oc, ic, nf, no);
      }
      // aggregate over dst, scale by 1/deg, add self-interaction
      zero(aggb, NN*oc*no);
      k_scatter<<<(NE*oc*no+255)/256, 256, 0, stream>>>(msgb, ei + NE, aggb, oc, no);
      bool hasSelf = first ? (dd == 0) : true;
      int icin = first ? 32 : 16;
      const float* fin = first ? hin : featIn[dd];
      k_finish<<<(NN*oc*no+255)/256, 256, 0, stream>>>(
          aggb, deg, hasSelf ? smat[li][dd] : (const float*)0, fin, outBuf[dd], oc, no, icin);
    }
    if(!last){
      for(int d = 0; d < 4; ++d)
        k_gnorm<<<(NN+63)/64, 64, 0, stream>>>(outBuf[d], nG[li][d], nB[li][d], nW[li][d], nC[li][d], 2*d+1);
      for(int d = 0; d < 4; ++d) featIn[d] = outBuf[d];
    } else {
      // WMMA output head: feats0 (NN x 64, m-dim == 1) @ Wmap (64 x 16)
      k_cvt<<<(NN*64+255)/256, 256, 0, stream>>>(outBuf[0], h1, NN*64);
      k_cvt<<<4, 256, 0, stream>>>(Wmap, W2h, 1024);
      k_head_h<<<NN/16, 32, 0, stream>>>(h1, W2h, bmap, (float*)d_out);
      k_head_x<<<(NN*3+255)/256, 256, 0, stream>>>(outBuf[1], x, (float*)d_out + NN*16);
    }
  }
}
